// LowRankMLSTMCell_84980222919432
// MI455X (gfx1250) — matrix operand order repaired and checked by reading, not compiled, not run on hardware
//
#include <hip/hip_runtime.h>

typedef __attribute__((ext_vector_type(16))) __bf16 v16bf;
typedef __attribute__((ext_vector_type(8)))  float  v8f;
typedef __attribute__((ext_vector_type(4)))  unsigned int u32x4;
typedef __attribute__((ext_vector_type(8)))  int i32x8;
typedef __attribute__((ext_vector_type(4)))  int i32x4;

constexpr int kB    = 32768;
constexpr int kIn   = 1024;
constexpr int kHid  = 1024;
constexpr int kNout = 3200;   // 3*1024 (q,k,v) + 4 (Wi) + 4 (Wf) + 120 pad -> 25 tiles of 128
constexpr int kMT   = 256;    // block M tile
constexpr int kNT   = 128;    // block N tile
constexpr int kKT   = 64;     // K slab per stage
constexpr int kLdsStride = 72;                       // halves per row (64 + 8 pad)
constexpr int kAHalves   = kMT * kLdsStride;         // 18432 halves
constexpr int kBHalves   = kNT * kLdsStride;         // 9216 halves
constexpr int kStageHalves = kAHalves + kBHalves;    // 27648 halves = 55296 B
constexpr int kLdsBytes  = 2 * kStageHalves * 2;     // double buffered = 110592 B

#if defined(__gfx1250__) && __has_builtin(__builtin_amdgcn_tensor_load_to_lds)
#define USE_TDM 1
#else
#define USE_TDM 0
#endif

__device__ __forceinline__ unsigned short f2bf(float f) {
  unsigned u = __builtin_bit_cast(unsigned, f);
  unsigned r = u + 0x7fffu + ((u >> 16) & 1u);   // round-to-nearest-even
  return (unsigned short)(r >> 16);
}

// ---------------- conversion kernels ----------------
__global__ __launch_bounds__(256) void cvt_x_kernel(const float* __restrict__ x,
                                                    unsigned short* __restrict__ xb) {
  int i = (blockIdx.x * 256 + threadIdx.x) * 4;
  float4 f = *(const float4*)(x + i);
  *(ushort4*)(xb + i) = make_ushort4(f2bf(f.x), f2bf(f.y), f2bf(f.z), f2bf(f.w));
}

__global__ __launch_bounds__(256) void cvt_w_kernel(
    const float* __restrict__ Wq, const float* __restrict__ Wk,
    const float* __restrict__ Wv, const float* __restrict__ Wi,
    const float* __restrict__ Wf, unsigned short* __restrict__ wb) {
  int i = (blockIdx.x * 256 + threadIdx.x) * 4;
  int row = i >> 10, col = i & 1023;
  const float* src = nullptr;
  if      (row < 1024) src = Wq + (size_t)row * 1024;
  else if (row < 2048) src = Wk + (size_t)(row - 1024) * 1024;
  else if (row < 3072) src = Wv + (size_t)(row - 2048) * 1024;
  else if (row < 3076) src = Wi + (size_t)(row - 3072) * 1024;
  else if (row < 3080) src = Wf + (size_t)(row - 3076) * 1024;
  ushort4 o;
  if (src) {
    float4 f = *(const float4*)(src + col);
    o = make_ushort4(f2bf(f.x), f2bf(f.y), f2bf(f.z), f2bf(f.w));
  } else {
    o = make_ushort4(0, 0, 0, 0);
  }
  *(ushort4*)(wb + i) = o;
}

// ---------------- TDM helper: async 2-D bf16 tile load into LDS ----------------
#if USE_TDM
// Loads a tile of (tile_rows x 64) bf16 elements from a row-major tensor
// (row stride kIn elements) starting at gptr, into LDS at byte offset
// lds_byte_off, inserting 4 DWORDs of padding after every 32 DWORDs
// (=> LDS row stride of 72 halves, matching the compute layout).
__device__ __forceinline__ void tdm_load_tile(unsigned lds_byte_off,
                                              const unsigned short* gptr,
                                              unsigned tensor_rows,
                                              unsigned tile_rows) {
  unsigned long long ga = (unsigned long long)gptr;
  u32x4 g0;
  g0[0] = 1u;                                             // count=1 (valid), user mode
  g0[1] = lds_byte_off;                                   // lds_addr
  g0[2] = (unsigned)(ga & 0xffffffffu);                   // global_addr[31:0]
  g0[3] = (unsigned)((ga >> 32) & 0x01ffffffu) | (2u << 30);  // addr[56:32] | type=2
  i32x8 g1;
  // data_size=2B, pad_enable, pad_interval=32 DWORDs (code 4), pad_amount=4 DWORDs (code 3)
  g1[0] = (int)((1u << 16) | (1u << 20) | (4u << 22) | (3u << 25));
  g1[1] = (int)((unsigned)kIn << 16);                     // tensor_dim0 = 1024 (bits 63:48)
  g1[2] = (int)((tensor_rows & 0xffffu) << 16);           // tensor_dim1[15:0] (bits 95:80)
  g1[3] = (int)(((tensor_rows >> 16) & 0xffffu) |         // tensor_dim1[31:16] (bits 111:96)
                ((unsigned)kKT << 16));                   // tile_dim0 = 64 (bits 127:112)
  g1[4] = (int)tile_rows;                                 // tile_dim1 (bits 143:128)
  g1[5] = kIn;                                            // tensor_dim0_stride[31:0] = 1024
  g1[6] = 0;                                              // stride hi / dim1_stride lo
  g1[7] = 0;
  i32x4 z4 = {0, 0, 0, 0};
  i32x8 z8 = {0, 0, 0, 0, 0, 0, 0, 0};
  __builtin_amdgcn_tensor_load_to_lds(g0, g1, z4, z4, z8, 0);
}
#endif

// ---------------- WMMA GEMM: Y[B,3200] = Xbf16 @ Wbf16^T ----------------
// Block tile 256x128, 8 waves, each wave computes 64x64 = 4x4 WMMA tiles.
extern __shared__ __align__(16) unsigned short smem_us[];

__global__ __launch_bounds__(256) void gemm_qkv_kernel(
    const unsigned short* __restrict__ X,   // [kB, kIn] bf16 bits
    const unsigned short* __restrict__ W,   // [kNout, kIn] bf16 bits
    float* __restrict__ Y) {                // [kB, kNout] f32
  const int tid  = threadIdx.x;
  const int lane = tid & 31;
  const int wid  = tid >> 5;
  const int wr   = wid >> 1;              // 0..3 -> 64-row slice
  const int wc   = wid & 1;               // 0..1 -> 64-col slice
  const int m0   = blockIdx.x * kMT;
  const int n0   = blockIdx.y * kNT;
  const int l15  = lane & 15;
  const int hiA  = (lane >> 4) * 8;       // A: K half select (0/8); also C/D M offset
  const int hiB  = (lane >> 4) * 16;      // unused; B reads select K half with hiA (0/8)

  const v8f vzero = {};
  v8f acc[4][4];
#pragma unroll
  for (int i = 0; i < 4; ++i)
#pragma unroll
    for (int j = 0; j < 4; ++j) acc[i][j] = vzero;

#if USE_TDM
  if (wid == 0) {
    tdm_load_tile(0u, X + (size_t)m0 * kIn, (unsigned)kB, (unsigned)kMT);
    tdm_load_tile((unsigned)(kAHalves * 2), W + (size_t)n0 * kIn, (unsigned)kNout,
                  (unsigned)kNT);
  }
#endif

  int ib = 0;
  for (int kt = 0; kt < kIn; kt += kKT, ib ^= 1) {
    unsigned short* As = smem_us + ib * kStageHalves;
    unsigned short* Bs = As + kAHalves;

#if USE_TDM
    if (wid == 0) {
      if (kt + kKT < kIn) {
        unsigned nb = (unsigned)((ib ^ 1) * kStageHalves * 2);
        tdm_load_tile(nb, X + (size_t)m0 * kIn + kt + kKT, (unsigned)kB, (unsigned)kMT);
        tdm_load_tile(nb + (unsigned)(kAHalves * 2), W + (size_t)n0 * kIn + kt + kKT,
                      (unsigned)kNout, (unsigned)kNT);
        __builtin_amdgcn_s_wait_tensorcnt(2);   // current stage's pair is done
      } else {
        __builtin_amdgcn_s_wait_tensorcnt(0);
      }
    }
#else
    // synchronous staging fallback: A 2048 16B-chunks, B 1024 16B-chunks
#pragma unroll
    for (int i = 0; i < 8; ++i) {
      int c = tid + i * 256;              // 0..2047
      int r = c >> 3, kg = (c & 7) << 3;
      *(uint4*)&As[r * kLdsStride + kg] =
          *(const uint4*)&X[(size_t)(m0 + r) * kIn + kt + kg];
    }
#pragma unroll
    for (int i = 0; i < 4; ++i) {
      int c = tid + i * 256;              // 0..1023
      int r = c >> 3, kg = (c & 7) << 3;
      *(uint4*)&Bs[r * kLdsStride + kg] =
          *(const uint4*)&W[(size_t)(n0 + r) * kIn + kt + kg];
    }
#endif
    __syncthreads();

#pragma unroll
    for (int ks = 0; ks < 2; ++ks) {
      const int kb = ks * 32;
      union { v16bf v; uint4 q[2]; } bfr[4];
#pragma unroll
      for (int j = 0; j < 4; ++j) {
        const unsigned short* p = &Bs[(size_t)(wc * 64 + j * 16 + l15) * kLdsStride + kb + hiA];
        bfr[j].q[0] = *(const uint4*)p;
        bfr[j].q[1] = *(const uint4*)(p + 16);
      }
#pragma unroll
      for (int i = 0; i < 4; ++i) {
        union { v16bf v; uint4 q[2]; } afr;
        const unsigned short* p = &As[(size_t)(wr * 64 + i * 16 + l15) * kLdsStride + kb + hiA];
        afr.q[0] = *(const uint4*)p;
        afr.q[1] = *(const uint4*)(p + 16);
#pragma unroll
        for (int j = 0; j < 4; ++j)
          acc[i][j] = __builtin_amdgcn_wmma_f32_16x16x32_bf16(
              false, afr.v, false, bfr[j].v, (short)0, acc[i][j], false, false);
      }
    }
    __syncthreads();
  }

  // C/D layout: VGPR p -> row M = p + (lane<16 ? 0 : 8), col N = lane&15
#pragma unroll
  for (int i = 0; i < 4; ++i) {
    const int rowb = m0 + wr * 64 + i * 16 + hiA;
#pragma unroll
    for (int j = 0; j < 4; ++j) {
      const int col = n0 + wc * 64 + j * 16 + l15;
      if (col < 3080) {                   // skip pad columns
#pragma unroll
        for (int p = 0; p < 8; ++p)
          Y[(size_t)(rowb + p) * kNout + col] = acc[i][j][p];
      }
    }
  }
}

// ---------------- epilogue: gates, rank-2 state update, h, LayerNorm ----------------
__global__ __launch_bounds__(256) void epilogue_kernel(
    const float* __restrict__ Y, const float* __restrict__ state,
    const float* __restrict__ bq, const float* __restrict__ bk,
    const float* __restrict__ bv, const float* __restrict__ U,
    const float* __restrict__ V,  const float* __restrict__ bi,
    const float* __restrict__ bfg, const float* __restrict__ gamma,
    const float* __restrict__ beta, float* __restrict__ out) {
  const int b = blockIdx.x;
  const int t = threadIdx.x;
  const int n = t >> 6;              // head (64 threads per head)
  const int d0 = (t & 63) * 4;       // in-head element base

  __shared__ float part[6][256];     // vc0 vc1 kc0 kc1 qc0 qc1 partials
  __shared__ float hcq[4][6];
  __shared__ float hc[4][2];
  __shared__ float red[256];

  const float* Yr = Y + (size_t)b * kNout;

  float q[4], k[4], v[4];
  float pv0 = 0, pv1 = 0, pk0 = 0, pk1 = 0, pq0 = 0, pq1 = 0;
#pragma unroll
  for (int e = 0; e < 4; ++e) {
    int j = 4 * t + e;               // 0..1023
    int d = d0 + e;                  // 0..255 in head
    q[e] = Yr[j]        + bq[j];
    k[e] = Yr[1024 + j] + bk[j];
    v[e] = Yr[2048 + j] + bv[j];
    float u0 = U[n * 512 + d * 2 + 0], u1 = U[n * 512 + d * 2 + 1];
    float w0 = V[n * 512 + d * 2 + 0], w1 = V[n * 512 + d * 2 + 1];
    pv0 += u0 * v[e]; pv1 += u1 * v[e];
    pk0 += w0 * k[e]; pk1 += w1 * k[e];
    pq0 += u0 * q[e]; pq1 += u1 * q[e];
  }
  part[0][t] = pv0; part[1][t] = pv1;
  part[2][t] = pk0; part[3][t] = pk1;
  part[4][t] = pq0; part[5][t] = pq1;
  __syncthreads();

  if (t < 24) {                      // 4 heads x 6 quantities
    int hn = t / 6, qi = t % 6;
    float s = 0.f;
    for (int j = 0; j < 64; ++j) s += part[qi][hn * 64 + j];
    hcq[hn][qi] = s;
  }
  __syncthreads();

  if (t < 16) {                      // C_new[b, hn, r, s] and diag * q_c
    int hn = t >> 2, r = (t >> 1) & 1, s = t & 1;
    float i_g = __expf(Yr[3072 + hn] + bi[hn]);
    float f_g = 1.f / (1.f + __expf(-(Yr[3076 + hn] + bfg[hn])));
    float c = f_g * state[(size_t)b * 16 + hn * 4 + r * 2 + s] +
              i_g * hcq[hn][r] * hcq[hn][2 + s];
    out[(size_t)kB * kHid + (size_t)b * 16 + hn * 4 + r * 2 + s] = c;
    if (r == s) hc[hn][r] = c * hcq[hn][4 + r];
  }
  __syncthreads();

  float h[4], s1 = 0.f, s2 = 0.f;
#pragma unroll
  for (int e = 0; e < 4; ++e) {
    int d = d0 + e;
    float hv = U[n * 512 + d * 2] * hc[n][0] + U[n * 512 + d * 2 + 1] * hc[n][1];
    h[e] = hv; s1 += hv; s2 += hv * hv;
  }
  red[t] = s1; __syncthreads();
  for (int st = 128; st > 0; st >>= 1) { if (t < st) red[t] += red[t + st]; __syncthreads(); }
  float mu = red[0] * (1.f / 1024.f);
  __syncthreads();
  red[t] = s2; __syncthreads();
  for (int st = 128; st > 0; st >>= 1) { if (t < st) red[t] += red[t + st]; __syncthreads(); }
  float var = red[0] * (1.f / 1024.f) - mu * mu;
  float inv = rsqrtf(var + 1e-5f);
#pragma unroll
  for (int e = 0; e < 4; ++e) {
    int j = 4 * t + e;
    out[(size_t)b * kHid + j] = (h[e] - mu) * inv * gamma[j] + beta[j];
  }
}

extern "C" void kernel_launch(void* const* d_in, const int* in_sizes, int n_in,
                              void* d_out, int out_size, void* d_ws, size_t ws_size,
                              hipStream_t stream) {
  (void)in_sizes; (void)n_in; (void)out_size; (void)ws_size;
  const float* x     = (const float*)d_in[0];
  const float* state = (const float*)d_in[1];
  const float* Wq    = (const float*)d_in[2];
  const float* bq    = (const float*)d_in[3];
  const float* Wk    = (const float*)d_in[4];
  const float* bk    = (const float*)d_in[5];
  const float* Wv    = (const float*)d_in[6];
  const float* bv    = (const float*)d_in[7];
  const float* U     = (const float*)d_in[8];
  const float* V     = (const float*)d_in[9];
  const float* Wi    = (const float*)d_in[10];
  const float* bi    = (const float*)d_in[11];
  const float* Wf    = (const float*)d_in[12];
  const float* bfg   = (const float*)d_in[13];
  const float* gamma = (const float*)d_in[14];
  const float* beta  = (const float*)d_in[15];
  float* out = (float*)d_out;

  char* ws = (char*)d_ws;
  unsigned short* xb = (unsigned short*)ws;                                  // 64 MB
  unsigned short* wb = (unsigned short*)(ws + (size_t)kB * kIn * 2);         // 6.5 MB
  float* Y = (float*)(ws + (size_t)kB * kIn * 2 + (size_t)kNout * kIn * 2);  // 419 MB

  cvt_x_kernel<<<(kB * kIn) / (256 * 4), 256, 0, stream>>>(x, xb);
  cvt_w_kernel<<<(kNout * kIn) / (256 * 4), 256, 0, stream>>>(Wq, Wk, Wv, Wi, Wf, wb);

  dim3 gg(kB / kMT, kNout / kNT);
  gemm_qkv_kernel<<<gg, 256, kLdsBytes, stream>>>(xb, wb, Y);

  epilogue_kernel<<<kB, 256, 0, stream>>>(Y, state, bq, bk, bv, U, V, bi, bfg,
                                          gamma, beta, out);
}
